// MultiHeadAttention_4612794875930
// MI455X (gfx1250) — matrix operand order repaired and checked by reading, not compiled, not run on hardware
//
#include <hip/hip_runtime.h>
#include <hip/hip_bf16.h>

// ---------------------------------------------------------------------------
// Multi-head attention for MI455X (gfx1250), bf16 WMMA everywhere.
//   D_MODEL=1024, H=16, DEPTH=64, B=2, S=2048
// Compute-bound (~70 GFLOP vs ~100MB traffic) => all matmuls through
// v_wmma_f32_16x16x32_bf16 (wave32, fp32 accum). All operands are L2-resident
// (192MB), so register-tiled WMMA with redundant L2 reads is fine.
//  - softmax scale (1/sqrt(64) * log2 e) folded into the Q projection epilogue
//  - softmax denominator on the matrix unit (P x ones -> 5th WMMA accumulator)
//  - 64-key flash chunks amortize the max-tree / rescale over 2x keys
// ---------------------------------------------------------------------------

#define D_MODEL   1024
#define NUM_HEADS 16
#define DEPTH     64
#define BATCH     2
#define SEQ       2048

typedef __bf16 bf16_t;
typedef __attribute__((ext_vector_type(16))) __bf16 v16bf;
typedef __attribute__((ext_vector_type(8)))  __bf16 v8bf;
typedef __attribute__((ext_vector_type(4)))  __bf16 v4bf;
typedef __attribute__((ext_vector_type(8)))  float  v8f;
typedef __attribute__((ext_vector_type(4)))  float  v4f;

static __device__ __forceinline__ v8f v8f_zero() {
  v8f z;
#pragma unroll
  for (int i = 0; i < 8; ++i) z[i] = 0.0f;
  return z;
}

static __device__ __forceinline__ float fast_exp2(float x) {
#if __has_builtin(__builtin_amdgcn_exp2f)
  return __builtin_amdgcn_exp2f(x);   // v_exp_f32 (hardware is base-2)
#else
  return exp2f(x);
#endif
}

static __device__ __forceinline__ float fast_rcp(float x) {
#if __has_builtin(__builtin_amdgcn_rcpf)
  return __builtin_amdgcn_rcpf(x);    // v_rcp_f32
#else
  return 1.0f / x;
#endif
}

static __device__ __forceinline__ v8f wmma_bf16(v16bf a, v16bf b, v8f c) {
  // 8 args: (neg_a, A, neg_b, B, c_mod, C, reuse_a, reuse_b)
  return __builtin_amdgcn_wmma_f32_16x16x32_bf16(false, a, false, b,
                                                 (short)0, c, false, false);
}

// A-matrix 16x32 bf16 fragment (k order shared with load_b_frag): lane L holds row m = L%16;
// 16 contiguous K values starting at k=(L/16)*16.
static __device__ __forceinline__ v16bf load_a_frag_bf16(const bf16_t* base,
                                                         int ld, int lane) {
  const bf16_t* row = base + (size_t)(lane & 15) * ld;
  const int seg0 = (lane >> 4) * 16;
  v8bf lo = *(const v8bf*)(row + seg0);
  v8bf hi = *(const v8bf*)(row + seg0 + 8);
  v16bf a;
#pragma unroll
  for (int i = 0; i < 8; ++i) { a[i] = lo[i]; a[i + 8] = hi[i]; }
  return a;
}

// B-matrix 32x16 bf16 fragment, source stored [N][K] row-major (torch Linear
// weight layout / Kh / Vt): lane L holds column n = L%16; 16 contiguous K
// values starting at k = (L/16)*16.
static __device__ __forceinline__ v16bf load_b_frag(const bf16_t* base,
                                                    int ld, int lane) {
  const bf16_t* row = base + (size_t)(lane & 15) * ld;
  const int seg = (lane >> 4) * 16;
  v8bf lo = *(const v8bf*)(row + seg);
  v8bf hi = *(const v8bf*)(row + seg + 8);
  v16bf b;
#pragma unroll
  for (int i = 0; i < 8; ++i) { b[i] = lo[i]; b[i + 8] = hi[i]; }
  return b;
}

// ---------------------------------------------------------------------------
// fp32 -> bf16 conversion, 4 elements/thread (b128 load, b64 store)
// ---------------------------------------------------------------------------
__global__ void cvt_f32_to_bf16_v4(const v4f* __restrict__ src,
                                   v4bf* __restrict__ dst, int n4) {
  int i = blockIdx.x * blockDim.x + threadIdx.x;
  if (i < n4) {
    v4f x = src[i];
    v4bf y;
#pragma unroll
    for (int j = 0; j < 4; ++j) y[j] = (bf16_t)x[j];
    dst[i] = y;
  }
}

// ---------------------------------------------------------------------------
// GEMM: Out[M,N] = (A[M,K] @ W[N,K]^T + bias[N]) * outScale, A/W bf16.
// One wave computes a 32(M) x 64(N) tile: 2 A-frags x 4 B-frags -> 8 WMMAs
// per 32-wide K chunk (B fragments shared across the two M sub-tiles).
// OMODE 0: fp32 row-major [M,N]                     (final dense -> d_out)
// OMODE 1: bf16 head-split [B,H,S,DEPTH]            (Q,K projections)
// OMODE 2: bf16 head-split transposed [B,H,DEPTH,S] (V projection)
// outScale folds the attention softmax scale into the Q projection.
// ---------------------------------------------------------------------------
template <int OMODE>
__global__ __launch_bounds__(128) void gemm_wmma_bf16(
    const bf16_t* __restrict__ A, const bf16_t* __restrict__ W,
    const float* __restrict__ bias, void* __restrict__ Out,
    int M, int N, int K, float outScale) {
  const int lane = threadIdx.x & 31;
  const int wave = blockIdx.x * (blockDim.x >> 5) + (threadIdx.x >> 5);
  const int mTiles = M >> 5;
  const int m0 = (wave % mTiles) << 5;
  const int n0 = (wave / mTiles) << 6;
  if (n0 >= N) return;  // grid sized exactly; never taken

  v8f acc[2][4];
#pragma unroll
  for (int mi = 0; mi < 2; ++mi)
#pragma unroll
    for (int t = 0; t < 4; ++t) acc[mi][t] = v8f_zero();

  for (int kc = 0; kc < K; kc += 32) {
    v16bf a0 = load_a_frag_bf16(A + (size_t)m0 * K + kc, K, lane);
    v16bf a1 = load_a_frag_bf16(A + (size_t)(m0 + 16) * K + kc, K, lane);
#pragma unroll
    for (int t = 0; t < 4; ++t) {
      v16bf b = load_b_frag(W + (size_t)(n0 + t * 16) * K + kc, K, lane);
      acc[0][t] = wmma_bf16(a0, b, acc[0][t]);
      acc[1][t] = wmma_bf16(a1, b, acc[1][t]);
    }
  }

  // C layout: VGPR r, lanes 0-15 -> (m0+r, n0+lane); lanes 16-31 -> (+8 rows)
  const int row_off = (lane >> 4) * 8;
  const int ncol = lane & 15;
#pragma unroll
  for (int mi = 0; mi < 2; ++mi) {
#pragma unroll
    for (int t = 0; t < 4; ++t) {
      const int n = n0 + t * 16 + ncol;
      const float bv = bias[n];
#pragma unroll
      for (int r = 0; r < 8; ++r) {
        const int m = m0 + mi * 16 + r + row_off;
        const float val = (acc[mi][t][r] + bv) * outScale;
        if constexpr (OMODE == 0) {
          ((float*)Out)[(size_t)m * N + n] = val;
        } else {
          const int bb = m >> 11, ss = m & (SEQ - 1);    // m -> (b, s)
          const int hh = n >> 6, dd = n & (DEPTH - 1);   // n -> (h, d)
          if constexpr (OMODE == 1)
            ((bf16_t*)Out)[((((size_t)bb * NUM_HEADS + hh) * SEQ + ss) << 6) +
                           dd] = (bf16_t)val;
          else
            ((bf16_t*)Out)[(((size_t)bb * NUM_HEADS + hh) * DEPTH + dd) * SEQ +
                           ss] = (bf16_t)val;
        }
      }
    }
  }
}

// ---------------------------------------------------------------------------
// One 64-key chunk of causal flash attention for a 16-query block.
// Q is pre-scaled by (1/sqrt(DEPTH))*log2(e), so raw WMMA scores are already
// in the exp2 domain. MASKED=true only for the single diagonal chunk.
// Denominator accumulates on the matrix unit: accL += P x ones, rescaled by
// the same alpha as the AV accumulators => accL[r] == online-softmax l.
// ---------------------------------------------------------------------------
template <bool MASKED>
static __device__ __forceinline__ void attn_chunk64(
    int kc, int q0, int lane, const bf16_t* __restrict__ Kbase,
    const bf16_t* __restrict__ Vbase, const v16bf& qf0, const v16bf& qf1,
    const v16bf& ones, bf16_t* ldsSlab, v8f (&acc)[4], v8f& accL,
    float (&mrow)[8]) {
  const int row_off = (lane >> 4) * 8;
  const int col = lane & 15;

  // --- 4 score tiles for keys [kc, kc+64) ---
  v8f s[4];
#pragma unroll
  for (int j = 0; j < 4; ++j) {
    s[j] = v8f_zero();
    v16bf kb = load_b_frag(Kbase + (size_t)(kc + j * 16) * DEPTH, DEPTH, lane);
    s[j] = wmma_bf16(qf0, kb, s[j]);
    kb = load_b_frag(Kbase + (size_t)(kc + j * 16) * DEPTH + 32, DEPTH, lane);
    s[j] = wmma_bf16(qf1, kb, s[j]);
  }

#pragma unroll
  for (int r = 0; r < 8; ++r) {
    float v[4];
#pragma unroll
    for (int j = 0; j < 4; ++j) v[j] = s[j][r];
    if constexpr (MASKED) {
      const int q = q0 + r + row_off;
#pragma unroll
      for (int j = 0; j < 4; ++j)
        if (kc + j * 16 + col > q) v[j] = -1e9f;
    }
    // lane-consistent row max (4-stage xor tree over the 16-lane half)
    float mx = fmaxf(fmaxf(v[0], v[1]), fmaxf(v[2], v[3]));
    mx = fmaxf(mx, __shfl_xor(mx, 1, 32));
    mx = fmaxf(mx, __shfl_xor(mx, 2, 32));
    mx = fmaxf(mx, __shfl_xor(mx, 4, 32));
    mx = fmaxf(mx, __shfl_xor(mx, 8, 32));
    const float mnew = fmaxf(mrow[r], mx);
    const float alpha = fast_exp2(mrow[r] - mnew);
    mrow[r] = mnew;
#pragma unroll
    for (int t = 0; t < 4; ++t) acc[t][r] *= alpha;
    accL[r] *= alpha;
    // stage P: C layout -> row-major 16x64 in this wave's private LDS slab
#pragma unroll
    for (int j = 0; j < 4; ++j)
      ldsSlab[(r + row_off) * 64 + j * 16 + col] =
          (bf16_t)fast_exp2(v[j] - mnew);
  }
  asm volatile("s_wait_dscnt 0x0" ::: "memory");  // lane-crossing LDS reads

  // --- P(16x64) x V(64x64) + P x ones (denominator), two 32-k halves ---
#pragma unroll
  for (int half = 0; half < 2; ++half) {
    v16bf pa = load_a_frag_bf16(ldsSlab + half * 32, 64, lane);
#pragma unroll
    for (int t = 0; t < 4; ++t) {
      v16bf vb = load_b_frag(Vbase + (size_t)(t * 16) * SEQ + kc + half * 32,
                             SEQ, lane);
      acc[t] = wmma_bf16(pa, vb, acc[t]);
    }
    accL = wmma_bf16(pa, ones, accL);
  }
}

// ---------------------------------------------------------------------------
// Causal flash attention. One wave owns one 16-query block of one (b,h).
// Output written bf16 in concat layout [B, S, H*DEPTH] for the dense GEMM.
// ---------------------------------------------------------------------------
__global__ __launch_bounds__(128) void flash_attn_wmma(
    const bf16_t* __restrict__ Qh, const bf16_t* __restrict__ Kh,
    const bf16_t* __restrict__ Vt, bf16_t* __restrict__ Out) {
  __shared__ bf16_t ldsP[4][16 * 64];  // private 16x64 P tile per wave (8KB)

  const int lane = threadIdx.x & 31;
  const int w = threadIdx.x >> 5;
  const int wave = blockIdx.x * 4 + w;
  const int qblk = wave & (SEQ / 16 - 1);
  const int h = (wave >> 7) & (NUM_HEADS - 1);
  const int b = wave >> 11;
  const int q0 = qblk << 4;

  const bf16_t* Qbase = Qh + (((size_t)b * NUM_HEADS + h) * SEQ + q0) * DEPTH;
  const bf16_t* Kbase = Kh + ((size_t)b * NUM_HEADS + h) * SEQ * DEPTH;
  const bf16_t* Vbase = Vt + ((size_t)b * NUM_HEADS + h) * DEPTH * SEQ;
  bf16_t* ldsSlab = &ldsP[w][0];

  // Q fragments: 16x64 as two 16x32 A fragments, loaded once (pre-scaled).
  v16bf qf0 = load_a_frag_bf16(Qbase + 0, DEPTH, lane);
  v16bf qf1 = load_a_frag_bf16(Qbase + 32, DEPTH, lane);

  // all-ones B fragment for the denominator WMMA
  v16bf ones;
#pragma unroll
  for (int i = 0; i < 16; ++i) ones[i] = (bf16_t)1.0f;

  v8f acc[4];
#pragma unroll
  for (int t = 0; t < 4; ++t) acc[t] = v8f_zero();
  v8f accL = v8f_zero();
  float mrow[8];
#pragma unroll
  for (int r = 0; r < 8; ++r) mrow[r] = -1e30f;

  // Bulk 64-key chunks (provably below the diagonal), then exactly one
  // masked diagonal chunk covering keys [kdiag, kdiag+64).
  const int kdiag = ((q0 + 15) >> 6) << 6;
  for (int kc = 0; kc < kdiag; kc += 64)
    attn_chunk64<false>(kc, q0, lane, Kbase, Vbase, qf0, qf1, ones, ldsSlab,
                        acc, accL, mrow);
  attn_chunk64<true>(kdiag, q0, lane, Kbase, Vbase, qf0, qf1, ones, ldsSlab,
                     acc, accL, mrow);

  // --- normalize (accL[r] == denominator l, replicated across lanes) and
  //     emit bf16 concat [B, S, H*DEPTH] ---
  const int row_off = (lane >> 4) * 8;
  const int col = lane & 15;
#pragma unroll
  for (int r = 0; r < 8; ++r) {
    const float inv = fast_rcp(accL[r]);
    const int q = q0 + r + row_off;
    bf16_t* orow = Out + ((size_t)b * SEQ + q) * D_MODEL + h * DEPTH;
#pragma unroll
    for (int t = 0; t < 4; ++t) orow[t * 16 + col] = (bf16_t)(acc[t][r] * inv);
  }
}

// ---------------------------------------------------------------------------
// Host-side orchestration (all on `stream`, graph-capture safe).
// ---------------------------------------------------------------------------
extern "C" void kernel_launch(void* const* d_in, const int* in_sizes, int n_in,
                              void* d_out, int out_size, void* d_ws,
                              size_t ws_size, hipStream_t stream) {
  const float* v_in = (const float*)d_in[0];
  const float* k_in = (const float*)d_in[1];
  const float* q_in = (const float*)d_in[2];
  // d_in[3] = additive causal mask; applied analytically in flash_attn_wmma.
  const float* wq_w = (const float*)d_in[4];
  const float* wq_b = (const float*)d_in[5];
  const float* wk_w = (const float*)d_in[6];
  const float* wk_b = (const float*)d_in[7];
  const float* wv_w = (const float*)d_in[8];
  const float* wv_b = (const float*)d_in[9];
  const float* wd_w = (const float*)d_in[10];
  const float* wd_b = (const float*)d_in[11];

  char* ws = (char*)d_ws;
  size_t off = 0;
  auto wsalloc = [&](size_t bytes) -> void* {
    void* p = ws + off;
    off += (bytes + 255) & ~(size_t)255;
    return p;
  };
  const size_t WBYTES = (size_t)D_MODEL * D_MODEL * sizeof(bf16_t);
  const size_t ABYTES = (size_t)BATCH * SEQ * D_MODEL * sizeof(bf16_t);
  bf16_t* wq_bf = (bf16_t*)wsalloc(WBYTES);
  bf16_t* wk_bf = (bf16_t*)wsalloc(WBYTES);
  bf16_t* wv_bf = (bf16_t*)wsalloc(WBYTES);
  bf16_t* wd_bf = (bf16_t*)wsalloc(WBYTES);
  bf16_t* q_bf = (bf16_t*)wsalloc(ABYTES);  // [M,K] bf16 activations
  bf16_t* k_bf = (bf16_t*)wsalloc(ABYTES);
  bf16_t* v_bf = (bf16_t*)wsalloc(ABYTES);
  bf16_t* Qh = (bf16_t*)wsalloc(ABYTES);    // [B,H,S,DEPTH], pre-scaled
  bf16_t* Kh = (bf16_t*)wsalloc(ABYTES);    // [B,H,S,DEPTH]
  bf16_t* Vt = (bf16_t*)wsalloc(ABYTES);    // [B,H,DEPTH,S]
  bf16_t* Cc = (bf16_t*)wsalloc(ABYTES);    // concat [B,S,H*DEPTH]

  const int wn4 = D_MODEL * D_MODEL / 4;
  cvt_f32_to_bf16_v4<<<(wn4 + 255) / 256, 256, 0, stream>>>(
      (const v4f*)wq_w, (v4bf*)wq_bf, wn4);
  cvt_f32_to_bf16_v4<<<(wn4 + 255) / 256, 256, 0, stream>>>(
      (const v4f*)wk_w, (v4bf*)wk_bf, wn4);
  cvt_f32_to_bf16_v4<<<(wn4 + 255) / 256, 256, 0, stream>>>(
      (const v4f*)wv_w, (v4bf*)wv_bf, wn4);
  cvt_f32_to_bf16_v4<<<(wn4 + 255) / 256, 256, 0, stream>>>(
      (const v4f*)wd_w, (v4bf*)wd_bf, wn4);
  const int an4 = BATCH * SEQ * D_MODEL / 4;
  cvt_f32_to_bf16_v4<<<(an4 + 255) / 256, 256, 0, stream>>>(
      (const v4f*)q_in, (v4bf*)q_bf, an4);
  cvt_f32_to_bf16_v4<<<(an4 + 255) / 256, 256, 0, stream>>>(
      (const v4f*)k_in, (v4bf*)k_bf, an4);
  cvt_f32_to_bf16_v4<<<(an4 + 255) / 256, 256, 0, stream>>>(
      (const v4f*)v_in, (v4bf*)v_bf, an4);

  const int M = BATCH * SEQ;                          // 4096
  const int gemm_waves = (M / 32) * (D_MODEL / 64);   // 2048 waves
  dim3 gemm_grid(gemm_waves / 4), gemm_block(128);

  const float QSCALE = 0.1803368801111244f;  // (1/sqrt(DEPTH)) * log2(e)
  gemm_wmma_bf16<1><<<gemm_grid, gemm_block, 0, stream>>>(
      q_bf, wq_bf, wq_b, (void*)Qh, M, D_MODEL, D_MODEL, QSCALE);
  gemm_wmma_bf16<1><<<gemm_grid, gemm_block, 0, stream>>>(
      k_bf, wk_bf, wk_b, (void*)Kh, M, D_MODEL, D_MODEL, 1.0f);
  gemm_wmma_bf16<2><<<gemm_grid, gemm_block, 0, stream>>>(
      v_bf, wv_bf, wv_b, (void*)Vt, M, D_MODEL, D_MODEL, 1.0f);

  const int attn_waves = BATCH * NUM_HEADS * (SEQ / 16);  // 4096 waves
  flash_attn_wmma<<<attn_waves / 4, 128, 0, stream>>>(Qh, Kh, Vt, Cc);

  gemm_wmma_bf16<0><<<gemm_grid, gemm_block, 0, stream>>>(
      Cc, wd_bf, wd_b, d_out, M, D_MODEL, D_MODEL, 1.0f);
}
